// AttnBlock_30365418783159
// MI455X (gfx1250) — compile-verified, matrix operand order checked by reading, not run on hardware
//
#include <hip/hip_runtime.h>
#include <hip/hip_bf16.h>
#include <cstdint>

typedef __attribute__((ext_vector_type(16))) _Float16 v16h;
typedef __attribute__((ext_vector_type(8)))  _Float16 v8h;
typedef __attribute__((ext_vector_type(4)))  _Float16 v4h;
typedef __attribute__((ext_vector_type(8)))  float    v8f;
typedef __attribute__((ext_vector_type(4)))  float    v4f;

#define C_DIM 128
#define N_SP  4096            // H*W
#define B_DIM 8
#define BN    (B_DIM * N_SP)  // 32768
#define QK_SCALE 0.08838834764831845f  // 128^-0.5
#define KV_STEP 32
#define NITER (N_SP / KV_STEP)  // 128

// ---------------------------------------------------------------------------
// Fragment loader: rows of a row-major f16 matrix -> WMMA A-frag (16xK rows)
// or B-frag (columns of B == rows of the row-major "transposed" source).
// Lane L (lo=L%16, hi=L/16): elements e<8  -> col 8*hi + e
//                            elements e>=8 -> col 16 + 8*hi + (e-8)
// Two contiguous 16-byte loads per lane (global_load_b128 / ds_load_b128).
// ---------------------------------------------------------------------------
__device__ __forceinline__ v16h load_frag(const _Float16* base, int ld,
                                          int lo, int hi) {
  const _Float16* p = base + (size_t)lo * ld + 8 * hi;
  v8h a = *(const v8h*)p;
  v8h b = *(const v8h*)(p + 16);
  v16h r;
#pragma unroll
  for (int i = 0; i < 8; ++i) { r[i] = a[i]; r[i + 8] = b[i]; }
  return r;
}

__device__ __forceinline__ v8f wmma_f16(v16h a, v16h b, v8f c) {
  return __builtin_amdgcn_wmma_f32_16x16x32_f16(false, a, false, b,
                                                (short)0, c, false, false);
}

// Async global->LDS copy of 16 bytes per lane (ASYNCcnt-tracked).
// lds_off = low 32 bits of the generic shared pointer == LDS byte offset
// (aperture occupies bits [63:32] of the flat shared address).
__device__ __forceinline__ void async_copy16(unsigned lds_off, const void* gsrc) {
  asm volatile("global_load_async_to_lds_b128 %0, %1, off"
               :: "v"(lds_off), "v"(gsrc)
               : "memory");
}

// ---------------------------------------------------------------------------
// GroupNorm: one block per (batch, group). 32 groups, 4 channels/group,
// reduce over 4096 positions * 4 channels. Output h in f16 [BN][C].
// ---------------------------------------------------------------------------
__global__ void gn_kernel(const float* __restrict__ x,
                          const float* __restrict__ gscale,
                          const float* __restrict__ gbias,
                          _Float16* __restrict__ h) {
  int b = blockIdx.x >> 5;
  int g = blockIdx.x & 31;
  const float* xp = x + (size_t)b * N_SP * C_DIM + g * 4;
  float s = 0.f, s2 = 0.f;
  for (int n = threadIdx.x; n < N_SP; n += blockDim.x) {
    v4f v = *(const v4f*)(xp + (size_t)n * C_DIM);
#pragma unroll
    for (int j = 0; j < 4; ++j) { s += v[j]; s2 += v[j] * v[j]; }
  }
  __shared__ float rs[256], rs2[256];
  rs[threadIdx.x] = s; rs2[threadIdx.x] = s2;
  __syncthreads();
  for (int st = 128; st > 0; st >>= 1) {
    if ((int)threadIdx.x < st) {
      rs[threadIdx.x]  += rs[threadIdx.x + st];
      rs2[threadIdx.x] += rs2[threadIdx.x + st];
    }
    __syncthreads();
  }
  float mean = rs[0] * (1.f / 16384.f);
  float var  = rs2[0] * (1.f / 16384.f) - mean * mean;
  float rstd = rsqrtf(var + 1e-6f);
  float sc[4], bi[4];
#pragma unroll
  for (int j = 0; j < 4; ++j) {
    float gsv = gscale[g * 4 + j];
    sc[j] = gsv * rstd;
    bi[j] = gbias[g * 4 + j] - mean * rstd * gsv;
  }
  _Float16* hp = h + (size_t)b * N_SP * C_DIM + g * 4;
  for (int n = threadIdx.x; n < N_SP; n += blockDim.x) {
    v4f v = *(const v4f*)(xp + (size_t)n * C_DIM);
    v4h o;
#pragma unroll
    for (int j = 0; j < 4; ++j) o[j] = (_Float16)(v[j] * sc[j] + bi[j]);
    *(v4h*)(hp + (size_t)n * C_DIM) = o;
  }
}

// ---------------------------------------------------------------------------
// Weight convert: 4 matrices [Cin][Cout] f32 -> transposed [Cout][Cin] f16.
// ---------------------------------------------------------------------------
__global__ void wcvt_kernel(const float* __restrict__ wq,
                            const float* __restrict__ wk,
                            const float* __restrict__ wv,
                            const float* __restrict__ wo,
                            _Float16* __restrict__ wT) {
  int idx = blockIdx.x * blockDim.x + threadIdx.x;   // 0 .. 65535
  int m = idx >> 14;
  int r = idx & 16383;
  int c = r >> 7;       // out channel (row of wT)
  int kk = r & 127;     // in channel
  const float* w = (m == 0) ? wq : (m == 1) ? wk : (m == 2) ? wv : wo;
  wT[idx] = (_Float16)w[kk * C_DIM + c];
}

// ---------------------------------------------------------------------------
// QKV projection: D(c-tile, n-tile) = wT_rows x h_rows, K=128 in 4 WMMA.
// grid (BN/16, 2, 3), block 128 (4 waves); wave -> c-tile blockIdx.y*4+wid.
// q,k stored row-major [BN][C] f16 (q pre-scaled by C^-0.5);
// v stored transposed [B][C][N] f16.
// ---------------------------------------------------------------------------
__global__ void qkv_kernel(const _Float16* __restrict__ h,
                           const _Float16* __restrict__ wT,
                           const float* __restrict__ bq,
                           const float* __restrict__ bk,
                           const float* __restrict__ bv,
                           _Float16* __restrict__ q,
                           _Float16* __restrict__ k,
                           _Float16* __restrict__ vT) {
  int wid  = threadIdx.x >> 5;
  int lane = threadIdx.x & 31, lo = lane & 15, hi = lane >> 4;
  int proj = blockIdx.z;
  int cb   = (blockIdx.y * 4 + wid) * 16;
  int n0   = blockIdx.x * 16;
  const _Float16* wt = wT + (size_t)proj * (C_DIM * C_DIM);
  const float* bias = (proj == 0) ? bq : (proj == 1) ? bk : bv;

  v8f acc = {};
#pragma unroll
  for (int j = 0; j < 4; ++j) {
    v16h A  = load_frag(wt + (size_t)cb * C_DIM + 32 * j, C_DIM, lo, hi);
    v16h Bf = load_frag(h + (size_t)n0 * C_DIM + 32 * j, C_DIM, lo, hi);
    acc = wmma_f16(A, Bf, acc);
  }
  // lane: position n = n0+lo ; channels c = cb + 8*hi + r
  int cch = cb + 8 * hi;
  if (proj < 2) {
    float scale = (proj == 0) ? QK_SCALE : 1.0f;
    v8h out;
#pragma unroll
    for (int r = 0; r < 8; ++r)
      out[r] = (_Float16)((acc[r] + bias[cch + r]) * scale);
    _Float16* dst = ((proj == 0) ? q : k) + (size_t)(n0 + lo) * C_DIM + cch;
    *(v8h*)dst = out;
  } else {
    int ng = n0 + lo, b = ng >> 12, n = ng & (N_SP - 1);
    _Float16* dst = vT + ((size_t)b * C_DIM + cch) * N_SP + n;
#pragma unroll
    for (int r = 0; r < 8; ++r)
      dst[(size_t)r * N_SP] = (_Float16)(acc[r] + bias[cch + r]);
  }
}

// ---------------------------------------------------------------------------
// Flash attention. Block = 4 waves, each wave owns 32 queries (128 q / block).
// K tile (32x128) and V^T tile (128x32) are staged cooperatively into LDS
// with async global->LDS b128 copies (double-buffered, ASYNCcnt),
// then fragments are read back via ds_load_b128.
// Scores computed transposed (S' = K x Q^T): the f32 D-fragment converts
// in-register into the B-fragment of O^T = V^T x P (no cross-lane movement).
// grid (N/128, B), block 128.
// ---------------------------------------------------------------------------
__device__ __forceinline__ void attn_step(const _Float16* bK, const _Float16* bV,
                                          const v16h (&Qf)[2][4],
                                          v8f (&oacc)[2][8],
                                          float (&m_run)[2], float (&l_run)[2],
                                          int lo, int hi) {
  v8f s[2][2];
#pragma unroll
  for (int st = 0; st < 2; ++st) {
    v16h Ka[4];
#pragma unroll
    for (int j = 0; j < 4; ++j)
      Ka[j] = load_frag(bK + st * 16 * C_DIM + 32 * j, C_DIM, lo, hi);
#pragma unroll
    for (int hq = 0; hq < 2; ++hq) {
      v8f acc = {};
#pragma unroll
      for (int j = 0; j < 4; ++j) acc = wmma_f16(Ka[j], Qf[hq][j], acc);
      s[hq][st] = acc;
    }
  }
  v16h pB[2];
#pragma unroll
  for (int hq = 0; hq < 2; ++hq) {
    float mloc = -1e30f;
#pragma unroll
    for (int r = 0; r < 8; ++r)
      mloc = fmaxf(mloc, fmaxf(s[hq][0][r], s[hq][1][r]));
    mloc = fmaxf(mloc, __shfl_xor(mloc, 16, 32));
    float m_new = fmaxf(m_run[hq], mloc);
    float alpha = __expf(m_run[hq] - m_new);
    float ssum = 0.f;
#pragma unroll
    for (int r = 0; r < 8; ++r) {
      float p = __expf(s[hq][0][r] - m_new); ssum += p; pB[hq][r] = (_Float16)p;
    }
#pragma unroll
    for (int r = 0; r < 8; ++r) {
      float p = __expf(s[hq][1][r] - m_new); ssum += p; pB[hq][r + 8] = (_Float16)p;
    }
    ssum += __shfl_xor(ssum, 16, 32);
    l_run[hq] = l_run[hq] * alpha + ssum;
    m_run[hq] = m_new;
#pragma unroll
    for (int t = 0; t < 8; ++t)
#pragma unroll
      for (int r = 0; r < 8; ++r) oacc[hq][t][r] *= alpha;
  }
#pragma unroll
  for (int t = 0; t < 8; ++t) {
    v16h Va = load_frag(bV + t * 16 * KV_STEP, KV_STEP, lo, hi);
#pragma unroll
    for (int hq = 0; hq < 2; ++hq)
      oacc[hq][t] = wmma_f16(Va, pB[hq], oacc[hq][t]);
  }
}

__global__ __launch_bounds__(128, 1)
void attn_kernel(const _Float16* __restrict__ q,
                 const _Float16* __restrict__ k,
                 const _Float16* __restrict__ vT,
                 _Float16* __restrict__ o) {
  int tix  = threadIdx.x;
  int wid  = tix >> 5;
  int lane = tix & 31, lo = lane & 15, hi = lane >> 4;
  int b  = blockIdx.y;
  int q0 = blockIdx.x * 128 + wid * 32;

  const _Float16* qp = q + ((size_t)b * N_SP + q0) * C_DIM;
  const _Float16* kb = k + (size_t)b * N_SP * C_DIM;
  const _Float16* vb = vT + (size_t)b * C_DIM * N_SP;

  // [buf][ K tile: 32x128 halves | V^T tile: 128x32 halves ]
  __shared__ __align__(16) _Float16 smem[2][8192];
  unsigned ldsK[2], ldsV[2];
#pragma unroll
  for (int i2 = 0; i2 < 2; ++i2) {
    ldsK[i2] = (unsigned)(uintptr_t)&smem[i2][0];
    ldsV[i2] = (unsigned)(uintptr_t)&smem[i2][4096];
  }

  v16h Qf[2][4];
#pragma unroll
  for (int hq = 0; hq < 2; ++hq)
#pragma unroll
    for (int j = 0; j < 4; ++j)
      Qf[hq][j] = load_frag(qp + (size_t)hq * 16 * C_DIM + 32 * j, C_DIM, lo, hi);

  v8f oacc[2][8];
#pragma unroll
  for (int hq = 0; hq < 2; ++hq)
#pragma unroll
    for (int t = 0; t < 8; ++t) oacc[hq][t] = (v8f){};
  float m_run[2] = {-1e30f, -1e30f};
  float l_run[2] = {0.f, 0.f};

  // --- stage helper (8 async b128 per thread => wave ASYNCcnt += 8) ---
  auto stage = [&](int kv0, unsigned dK, unsigned dV) {
    const _Float16* gk = kb + (size_t)kv0 * C_DIM;   // contiguous 8KB
#pragma unroll
    for (int j = 0; j < 4; ++j) {
      int c = tix + j * 128;                         // chunk 0..511 (16B each)
      async_copy16(dK + c * 16, gk + c * 8);
    }
#pragma unroll
    for (int j = 0; j < 4; ++j) {
      int c = tix + j * 128;
      int row = c >> 2, off = (c & 3) * 8;           // V^T: 128 rows x 64B
      async_copy16(dV + (unsigned)(row * KV_STEP + off) * 2,
                   vb + (size_t)row * N_SP + kv0 + off);
    }
  };

  stage(0, ldsK[0], ldsV[0]);                        // prologue prefetch

  for (int it = 0; it < NITER - 1; ++it) {
    int cur = it & 1;
    __syncthreads();                                 // buf[cur^1] free to overwrite
    stage((it + 1) * KV_STEP, ldsK[cur ^ 1], ldsV[cur ^ 1]);
    asm volatile("s_wait_asynccnt 0x8" ::: "memory"); // buf[cur] writes retired
    __syncthreads();                                 // all waves' buf[cur] ready
    attn_step(&smem[cur][0], &smem[cur][4096], Qf, oacc, m_run, l_run, lo, hi);
  }
  {
    int cur = (NITER - 1) & 1;
    __syncthreads();
    asm volatile("s_wait_asynccnt 0x0" ::: "memory");
    __syncthreads();
    attn_step(&smem[cur][0], &smem[cur][4096], Qf, oacc, m_run, l_run, lo, hi);
  }

#pragma unroll
  for (int hq = 0; hq < 2; ++hq) {
    float inv_l = 1.0f / l_run[hq];
    _Float16* ob = o + ((size_t)b * N_SP + q0 + hq * 16 + lo) * C_DIM + 8 * hi;
#pragma unroll
    for (int t = 0; t < 8; ++t) {
      v8h out;
#pragma unroll
      for (int r = 0; r < 8; ++r) out[r] = (_Float16)(oacc[hq][t][r] * inv_l);
      *(v8h*)(ob + t * 16) = out;
    }
  }
}

// ---------------------------------------------------------------------------
// Output projection + bias + residual (f32 out).
// grid (BN/16, 2), block 128.
// ---------------------------------------------------------------------------
__global__ void proj_out_kernel(const _Float16* __restrict__ o,
                                const _Float16* __restrict__ woT,
                                const float* __restrict__ bo,
                                const float* __restrict__ x,
                                float* __restrict__ out) {
  int wid  = threadIdx.x >> 5;
  int lane = threadIdx.x & 31, lo = lane & 15, hi = lane >> 4;
  int cb = (blockIdx.y * 4 + wid) * 16;
  int n0 = blockIdx.x * 16;

  v8f acc = {};
#pragma unroll
  for (int j = 0; j < 4; ++j) {
    v16h A  = load_frag(woT + (size_t)cb * C_DIM + 32 * j, C_DIM, lo, hi);
    v16h Bf = load_frag(o + (size_t)n0 * C_DIM + 32 * j, C_DIM, lo, hi);
    acc = wmma_f16(A, Bf, acc);
  }
  int n = n0 + lo, cc = cb + 8 * hi;
  const float* xp = x + (size_t)n * C_DIM + cc;
  float* op = out + (size_t)n * C_DIM + cc;
  v4f r0 = *(const v4f*)xp;
  v4f r1 = *(const v4f*)(xp + 4);
#pragma unroll
  for (int r = 0; r < 4; ++r) r0[r] += acc[r] + bo[cc + r];
#pragma unroll
  for (int r = 0; r < 4; ++r) r1[r] += acc[r + 4] + bo[cc + 4 + r];
  *(v4f*)op = r0;
  *(v4f*)(op + 4) = r1;
}

// ---------------------------------------------------------------------------
extern "C" void kernel_launch(void* const* d_in, const int* in_sizes, int n_in,
                              void* d_out, int out_size, void* d_ws, size_t ws_size,
                              hipStream_t stream) {
  const float* x  = (const float*)d_in[0];
  const float* gs = (const float*)d_in[1];
  const float* gb = (const float*)d_in[2];
  const float* wq = (const float*)d_in[3];
  const float* bq = (const float*)d_in[4];
  const float* wk = (const float*)d_in[5];
  const float* bk = (const float*)d_in[6];
  const float* wv = (const float*)d_in[7];
  const float* bv = (const float*)d_in[8];
  const float* wo = (const float*)d_in[9];
  const float* bo = (const float*)d_in[10];
  float* out = (float*)d_out;

  const size_t elems = (size_t)BN * C_DIM;      // 4M halves per buffer
  _Float16* h  = (_Float16*)d_ws;
  _Float16* qb = h  + elems;
  _Float16* kb = qb + elems;
  _Float16* vT = kb + elems;
  _Float16* ob = vT + elems;
  _Float16* wT = ob + elems;                    // 4 * 128*128 halves

  gn_kernel<<<B_DIM * 32, 256, 0, stream>>>(x, gs, gb, h);
  wcvt_kernel<<<256, 256, 0, stream>>>(wq, wk, wv, wo, wT);
  qkv_kernel<<<dim3(BN / 16, 2, 3), 128, 0, stream>>>(h, wT, bq, bk, bv, qb, kb, vT);
  attn_kernel<<<dim3(N_SP / 128, B_DIM), 128, 0, stream>>>(qb, kb, vT, ob);
  proj_out_kernel<<<dim3(BN / 16, 2), 128, 0, stream>>>(
      ob, wT + 3 * (size_t)(C_DIM * C_DIM), bo, x, out);
}
